// PointTransformerLayer_47382079209947
// MI455X (gfx1250) — matrix operand order repaired and checked by reading, not compiled, not run on hardware
//
#include <hip/hip_runtime.h>
#include <hip/hip_bf16.h>
#include <math.h>

// ---------------------------------------------------------------------------
// Point Transformer layer for MI455X (gfx1250), wave32 + WMMA f16->f32.
// Inputs (setup_inputs order): x, pos, mask(bool->u8), Wqkv, Wp1, bp1, Wp2,
// bp2, Wa1, ba1, Wa2, ba2.  Output: agg (B,N,128) f32.
// ---------------------------------------------------------------------------

constexpr int Bn = 2;
constexpr int Nn = 2048;
constexpr int D  = 128;
constexpr int KN = 16;
constexpr int PH = 64;     // pos-mlp hidden
constexpr int AH = 512;    // attn-mlp hidden
constexpr int QD = 3 * D;  // 384
constexpr int BN = Bn * Nn;    // 4096 query rows
constexpr int R  = BN * KN;    // 65536 (b,n,k) rows

typedef __attribute__((ext_vector_type(16))) _Float16 v16h;
typedef __attribute__((ext_vector_type(8)))  _Float16 v8h;
typedef __attribute__((ext_vector_type(8)))  float    v8f;

// A-operand load: per lane, halves 0..15 = K[kk0..kk0+15] (contiguous,
// matching B); caller passes p already offset by (row*stride + kb + kk0),
// kk0 = 16*hi. One aligned 32B chunk.
__device__ __forceinline__ v16h load_a16(const _Float16* p) {
    return *(const v16h*)(p);
}

__device__ __forceinline__ v8f wmma_f16(v16h a, v16h b, v8f c) {
    // D = A(16x32 f16) * B(32x16 f16) + C(16x16 f32)
    return __builtin_amdgcn_wmma_f32_16x16x32_f16(
        /*neg_a=*/false, a, /*neg_b=*/false, b,
        /*c_mod=*/(short)0, c, /*reuse_a=*/false, /*reuse_b=*/false);
}

// ---------------------------------------------------------------------------
__global__ void cvt_f16_kernel(const float* __restrict__ s,
                               _Float16* __restrict__ d, int n) {
    int i = blockIdx.x * 256 + threadIdx.x;
    if (i < n) d[i] = (_Float16)s[i];
}

// ---------------------------------------------------------------------------
// qkv = x @ Wqkv^T : (4096 x 128) x (128 x 384). One wave per 16x16 tile.
__global__ void qkv_wmma_kernel(const _Float16* __restrict__ xh,
                                const _Float16* __restrict__ wqh,
                                float* __restrict__ qkv) {
    int gw   = (blockIdx.x * blockDim.x + threadIdx.x) >> 5;
    int lane = threadIdx.x & 31;
    int tm = gw / (QD / 16);          // 256 row tiles
    int tn = gw % (QD / 16);          // 24 col tiles
    int ln = lane & 15, hi = lane >> 4;

    const _Float16* ar = xh  + (size_t)(tm * 16 + ln) * D + hi * 16;
    const _Float16* br = wqh + (size_t)(tn * 16 + ln) * D + hi * 16;
    v8f acc = {};
#pragma unroll
    for (int kb = 0; kb < D; kb += 32) {
        v16h a = load_a16(ar + kb);
        v16h b = *(const v16h*)(br + kb);
        acc = wmma_f16(a, b, acc);
    }
    int col = tn * 16 + ln;
#pragma unroll
    for (int j = 0; j < 8; ++j) {
        int row = tm * 16 + j + hi * 8;      // C/D layout: lanes>=16 -> M+8
        qkv[(size_t)row * QD + col] = acc[j];
    }
}

// ---------------------------------------------------------------------------
// kNN top-16 by squared distance (monotone in true distance); thread/query.
// Top-16 list kept fully register-resident: constant-indexed unrolled
// insertion (cmov chain) guarded by the cheap register test dd < bd[15].
__global__ void knn_kernel(const float* __restrict__ pos,
                           const unsigned char* __restrict__ mask,
                           int* __restrict__ idx,
                           unsigned char* __restrict__ mg) {
    int t = blockIdx.x * blockDim.x + threadIdx.x;
    if (t >= BN) return;
    int b = t / Nn, n = t - b * Nn;
    const float* pb = pos + (size_t)b * Nn * 3;
    float px = pb[3 * n], py = pb[3 * n + 1], pz = pb[3 * n + 2];
    bool mq = mask[t] != 0;

    float bd[KN];
    int   bi[KN];
#pragma unroll
    for (int i = 0; i < KN; ++i) { bd[i] = 3.3e38f; bi[i] = i; }

    for (int j = 0; j < Nn; ++j) {
        float dx = px - pb[3 * j];
        float dy = py - pb[3 * j + 1];
        float dz = pz - pb[3 * j + 2];
        float d2 = dx * dx + dy * dy + dz * dz;
        bool ok  = mq && (mask[b * Nn + j] != 0);
        float dd = ok ? d2 : 3.2e38f;
        if (dd < bd[KN - 1]) {          // rare: ~K*ln(N) hits over the scan
            float nd = dd;
            int   ni = j;
#pragma unroll
            for (int i = 0; i < KN; ++i) {   // constant indices -> VGPRs
                bool  lt = nd < bd[i];       // strict: stable on ties
                float td = bd[i];
                int   ti = bi[i];
                bd[i] = lt ? nd : td;
                bi[i] = lt ? ni : ti;
                nd = lt ? td : nd;
                ni = lt ? ti : ni;
            }
        }
    }
#pragma unroll
    for (int i = 0; i < KN; ++i) {
        idx[t * KN + i] = bi[i];
        mg[t * KN + i]  = (mq && (mask[b * Nn + bi[i]] != 0)) ? 1 : 0;
    }
}

// ---------------------------------------------------------------------------
// pos-mlp layer 1: h = relu(Wp1 @ rel_pos + bp1), 64 outputs per (b,n,k) row.
__global__ void hpos_kernel(const float* __restrict__ pos,
                            const int* __restrict__ idx,
                            const float* __restrict__ Wp1,
                            const float* __restrict__ bp1,
                            _Float16* __restrict__ h) {
    int r = blockIdx.x;          // 0..R-1
    int t = threadIdx.x;         // 0..63
    int bn = r >> 4;
    int b = bn / Nn, n = bn - b * Nn;
    int nb = idx[r];
    const float* pb = pos + (size_t)b * Nn * 3;
    float rx = pb[3 * n]     - pb[3 * nb];
    float ry = pb[3 * n + 1] - pb[3 * nb + 1];
    float rz = pb[3 * n + 2] - pb[3 * nb + 2];
    float acc = fmaf(Wp1[3 * t], rx,
                fmaf(Wp1[3 * t + 1], ry,
                fmaf(Wp1[3 * t + 2], rz, bp1[t])));
    h[(size_t)r * PH + t] = (_Float16)fmaxf(acc, 0.0f);
}

// ---------------------------------------------------------------------------
// Fused: rel_pos_emb = h @ Wp2^T + bp2 (WMMA over K=64), then
//   attn_in = (q - k_g + emb) -> f16, v_eff = v_g + emb -> f32.
// One wave per (b,n); its 16 M-rows are exactly the K=16 neighbors.
// A tile is invariant across the 8 output tiles -> hoisted (2 loads total).
__global__ void emb_wmma_kernel(const _Float16* __restrict__ h,
                                const _Float16* __restrict__ wp2h,
                                const float* __restrict__ bp2,
                                const float* __restrict__ qkv,
                                const int* __restrict__ idx,
                                _Float16* __restrict__ attn_in,
                                float* __restrict__ veff) {
    int lane = threadIdx.x & 31, wid = threadIdx.x >> 5;
    int bn = blockIdx.x * 8 + wid;
    int b  = bn >> 11;                // / Nn
    int ln = lane & 15, hi = lane >> 4;

    const _Float16* ar = h + ((size_t)bn * KN + ln) * PH + hi * 16;
    v16h a0 = load_a16(ar);
    v16h a1 = load_a16(ar + 32);

    int nbs[8];
#pragma unroll
    for (int j = 0; j < 8; ++j) nbs[j] = idx[bn * KN + j + hi * 8];

#pragma unroll
    for (int tn = 0; tn < 8; ++tn) {
        int c = tn * 16 + ln;
        const _Float16* br = wp2h + (size_t)c * PH + hi * 16;
        v8f acc = {};
        acc = wmma_f16(a0, *(const v16h*)(br),      acc);
        acc = wmma_f16(a1, *(const v16h*)(br + 32), acc);

        float qv   = qkv[(size_t)bn * QD + c];
        float bias = bp2[c];
#pragma unroll
        for (int j = 0; j < 8; ++j) {
            int m = j + hi * 8;
            size_t r  = (size_t)bn * KN + m;
            size_t nr = ((size_t)b * Nn + nbs[j]) * QD;
            float emb = acc[j] + bias;
            float kk  = qkv[nr + D + c];
            float vv  = qkv[nr + 2 * D + c];
            attn_in[r * D + c] = (_Float16)(qv - kk + emb);
            veff[r * D + c]    = vv + emb;
        }
    }
}

// ---------------------------------------------------------------------------
// Fused attn MLP (128->512->128 via WMMA, activations staged in LDS) +
// softmax over the 16 neighbor rows + weighted aggregation of v_eff.
// One wave per (b,n); 4 waves / 128 threads per block; dyn LDS 80KB/WG.
__global__ void attn_wmma_kernel(const _Float16* __restrict__ attn_in,
                                 const float* __restrict__ veff,
                                 const _Float16* __restrict__ wa1h,
                                 const _Float16* __restrict__ wa2h,
                                 const float* __restrict__ ba1,
                                 const float* __restrict__ ba2,
                                 const unsigned char* __restrict__ mg,
                                 float* __restrict__ out) {
    extern __shared__ _Float16 smem[];
    int lane = threadIdx.x & 31, wid = threadIdx.x >> 5;
    int bn = blockIdx.x * 4 + wid;
    int ln = lane & 15, hi = lane >> 4;

    _Float16* ain = smem + (size_t)wid * (KN * D);                 // 16x128 f16
    _Float16* ab  = smem + 4 * (KN * D) + (size_t)wid * (KN * AH); // 16x512 f16

    // Pull the L2-resident weight tiles toward this WGP.
    __builtin_prefetch(wa1h + (size_t)lane * 64, 0, 0);
    __builtin_prefetch(wa2h + (size_t)lane * 64, 0, 0);

    // Stage the 16x128 attn_in tile into LDS (coalesced).
    {
        const _Float16* src = attn_in + (size_t)bn * KN * D;
        for (int i = lane; i < KN * D; i += 32) ain[i] = src[i];
    }

    // ---- Layer 1: a = relu(attn_in @ Wa1^T + ba1), streamed to LDS ----
    // A tile (16x128) invariant across all 32 output tiles -> hoist to regs.
    {
        const _Float16* arow = ain + ln * D + hi * 16;
        v16h A[4];
#pragma unroll
        for (int kk = 0; kk < 4; ++kk) A[kk] = load_a16(arow + kk * 32);

        for (int t1 = 0; t1 < AH / 16; ++t1) {
            int c1 = t1 * 16 + ln;
            const _Float16* br = wa1h + (size_t)c1 * D + hi * 16;
            v8f acc = {};
#pragma unroll
            for (int kk = 0; kk < 4; ++kk)
                acc = wmma_f16(A[kk], *(const v16h*)(br + kk * 32), acc);

            float bias = ba1[c1];
#pragma unroll
            for (int j = 0; j < 8; ++j) {
                int m = j + hi * 8;
                ab[m * AH + c1] = (_Float16)fmaxf(acc[j] + bias, 0.0f);
            }
        }
    }

    // ---- Layer 2: sim = a @ Wa2^T (+ba2), kept in registers ----
    v8f zero = {};
    v8f acc2[8];
#pragma unroll
    for (int t = 0; t < 8; ++t) acc2[t] = zero;

    const _Float16* arow2 = ab + ln * AH + hi * 16;
    for (int kb2 = 0; kb2 < AH; kb2 += 32) {
        v16h a = load_a16(arow2 + kb2);
#pragma unroll
        for (int t2 = 0; t2 < 8; ++t2) {
            const _Float16* br =
                wa2h + (size_t)(t2 * 16 + ln) * AH + hi * 16 + kb2;
            v16h bb = *(const v16h*)br;
            acc2[t2] = wmma_f16(a, bb, acc2[t2]);
        }
    }

    // ---- Mask, softmax over 16 neighbors, aggregate v_eff ----
    float masked[8];
#pragma unroll
    for (int j = 0; j < 8; ++j)
        masked[j] = mg[bn * KN + j + hi * 8] ? 0.0f : 1.0f;

#pragma unroll
    for (int t2 = 0; t2 < 8; ++t2) {
        int c = t2 * 16 + ln;
        float bias = ba2[c];
        float s[8];
        float mx = -3.4e38f;
#pragma unroll
        for (int j = 0; j < 8; ++j) {
            float v = acc2[t2][j] + bias;
            if (masked[j] > 0.5f) v = -3.4e38f;
            s[j] = v;
            mx = fmaxf(mx, v);
        }
        // column c spans lanes {c, c+16}: finish 16-row reduction via xor-16.
        mx = fmaxf(mx, __shfl_xor(mx, 16, 32));
        float sum = 0.0f, p = 0.0f;
#pragma unroll
        for (int j = 0; j < 8; ++j) {
            float e = __expf(s[j] - mx);
            sum += e;
            int m = j + hi * 8;
            p += e * veff[((size_t)bn * KN + m) * D + c];
        }
        sum += __shfl_xor(sum, 16, 32);
        p   += __shfl_xor(p, 16, 32);
        if (lane < 16) out[(size_t)bn * D + c] = p / sum;
    }
}

// ---------------------------------------------------------------------------
extern "C" void kernel_launch(void* const* d_in, const int* in_sizes, int n_in,
                              void* d_out, int out_size, void* d_ws,
                              size_t ws_size, hipStream_t stream) {
    (void)in_sizes; (void)n_in; (void)out_size; (void)ws_size;

    const float* x    = (const float*)d_in[0];
    const float* pos  = (const float*)d_in[1];
    const unsigned char* mask = (const unsigned char*)d_in[2]; // jnp bool = u8
    const float* Wqkv = (const float*)d_in[3];
    const float* Wp1  = (const float*)d_in[4];
    const float* bp1  = (const float*)d_in[5];
    const float* Wp2  = (const float*)d_in[6];
    const float* bp2  = (const float*)d_in[7];
    const float* Wa1  = (const float*)d_in[8];
    const float* ba1  = (const float*)d_in[9];
    const float* Wa2  = (const float*)d_in[10];
    const float* ba2  = (const float*)d_in[11];
    float* out = (float*)d_out;

    char* base = (char*)d_ws;
    size_t off = 0;
    auto alloc = [&](size_t bytes) -> void* {
        void* p = base + off;
        off = (off + bytes + 255) & ~(size_t)255;
        return p;
    };

    _Float16* xh    = (_Float16*)alloc((size_t)BN * D * 2);   //  1   MB
    _Float16* wqkvh = (_Float16*)alloc((size_t)QD * D * 2);   // 96   KB
    _Float16* wp2h  = (_Float16*)alloc((size_t)D * PH * 2);   // 16   KB
    _Float16* wa1h  = (_Float16*)alloc((size_t)AH * D * 2);   // 128  KB
    _Float16* wa2h  = (_Float16*)alloc((size_t)D * AH * 2);   // 128  KB
    float*    qkv   = (float*)   alloc((size_t)BN * QD * 4);  //  6   MB
    int*      idx   = (int*)     alloc((size_t)R * 4);        // 256  KB
    unsigned char* mgbuf = (unsigned char*)alloc((size_t)R);  //  64  KB
    _Float16* hbuf  = (_Float16*)alloc((size_t)R * PH * 2);   //  8   MB
    _Float16* ain   = (_Float16*)alloc((size_t)R * D * 2);    // 16   MB
    float*    veff  = (float*)   alloc((size_t)R * D * 4);    // 32   MB

    // f32 -> f16 conversions
    auto cvt = [&](const float* s, _Float16* d, int n) {
        cvt_f16_kernel<<<(n + 255) / 256, 256, 0, stream>>>(s, d, n);
    };
    cvt(x,    xh,    BN * D);
    cvt(Wqkv, wqkvh, QD * D);
    cvt(Wp2,  wp2h,  D * PH);
    cvt(Wa1,  wa1h,  AH * D);
    cvt(Wa2,  wa2h,  D * AH);

    // QKV GEMM: 256 x 24 tiles, 1 wave each, 8 waves/block.
    qkv_wmma_kernel<<<(256 * 24) / 8, 256, 0, stream>>>(xh, wqkvh, qkv);

    // kNN top-16
    knn_kernel<<<(BN + 255) / 256, 256, 0, stream>>>(pos, mask, idx, mgbuf);

    // pos-mlp layer 1
    hpos_kernel<<<R, PH, 0, stream>>>(pos, idx, Wp1, bp1, hbuf);

    // fused pos-emb WMMA + gather (1 wave per (b,n), 8 waves/block)
    emb_wmma_kernel<<<BN / 8, 256, 0, stream>>>(hbuf, wp2h, bp2, qkv, idx,
                                                ain, veff);

    // fused attn MLP + softmax + aggregation
    // dyn LDS: 4 waves * (16*128 + 16*512) halves * 2B = 81920 B (<< 320 KB)
    size_t lds = (size_t)4 * (KN * D + KN * AH) * sizeof(_Float16);
    attn_wmma_kernel<<<BN / 4, 128, lds, stream>>>(ain, veff, wa1h, wa2h,
                                                   ba1, ba2, mgbuf, out);
}
